// MultiScaleTransformerEncoder_31250182046548
// MI455X (gfx1250) — matrix operand order repaired and checked by reading, not compiled, not run on hardware
//
#include <hip/hip_runtime.h>

// ---------------------------------------------------------------------------
// MultiScaleTransformerEncoder forward for gfx1250 (MI455X, wave32, WMMA)
// ---------------------------------------------------------------------------

typedef __attribute__((ext_vector_type(16))) __bf16 bf16x16;
typedef __attribute__((ext_vector_type(8)))  float  f32x8;

#define B_DIM 64
#define T_DIM 64
#define P_DIM 25
#define CIN   256
#define CS    192     // SMALL
#define CL    64      // LARGE
#define JS    10
#define JL    15
#define GS    4096    // B*T groups
#define RS    (GS*JS) // 40960 small token rows
#define RL    (GS*JL) // 61440 large token rows
#define FFD   6400    // OUTPUT_DIM * NUM_POINT
#define FFS   768     // 4*CS
#define FFL   256     // 4*CL
#define HEADS 8

__constant__ int c_UP[15]   = {2,3,10,11,6,7,8,9,4,5,20,21,22,23,24};
__constant__ int c_DOWN[10] = {16,17,18,19,12,13,14,15,0,1};

__device__ __forceinline__ unsigned short f32_to_bf16(float f) {
    unsigned u = __float_as_uint(f);
    unsigned r = 0x7FFFu + ((u >> 16) & 1u);
    return (unsigned short)((u + r) >> 16);
}
__device__ __forceinline__ float bf16_to_f32(unsigned short h) {
    return __uint_as_float(((unsigned)h) << 16);
}

// ---------------------------------------------------------------------------
// WMMA GEMM:  C[M,N] = A[M,K](bf16) * W[N,K](bf16)^T  (+bias +res, opt GELU)
// M multiple of 128, K multiple of 64, N multiple of 16.
// Block 256 threads (8 waves), block tile 128x128, wave tile 64x32.
// K loop is unrolled x2 with double-buffered fragments so VMEM for the next
// half-tile overlaps the WMMA batch of the current one.
// Out-of-range weight rows are clamped to row 0 (their columns are never
// stored), avoiding any EXEC-mask divergence in the hot loop.
// ---------------------------------------------------------------------------
union Frag32B { uint4 u[2]; bf16x16 v; };

template <bool GELU>
__global__ __launch_bounds__(256)
void k_gemm_bf16(const unsigned short* __restrict__ A,
                 const unsigned short* __restrict__ W,
                 const float* __restrict__ bias,
                 const float* __restrict__ res,
                 float* __restrict__ Cf,
                 unsigned short* __restrict__ Cb,
                 int M, int N, int K)
{
    const int lane   = threadIdx.x & 31;
    const int wave   = threadIdx.x >> 5;
    const int laneLo = lane & 15;
    const int laneHi = lane >> 4;
    const int wm     = wave & 1;       // 2 waves along M
    const int wn     = wave >> 1;      // 4 waves along N
    const int mBase  = blockIdx.y * 128 + wm * 64;
    const int nBase  = blockIdx.x * 128 + wn * 32;

    // Per-row base pointers (K-invariant).
    const unsigned short* pa[4];
    #pragma unroll
    for (int mi = 0; mi < 4; ++mi)
        pa[mi] = A + (size_t)(mBase + mi * 16 + laneLo) * K + laneHi * 8;
    const unsigned short* pb[2];
    #pragma unroll
    for (int ni = 0; ni < 2; ++ni) {
        int wr = nBase + ni * 16 + laneLo;
        if (wr >= N) wr = 0;                    // clamp: column masked at store
        pb[ni] = W + (size_t)wr * K + laneHi * 8;
    }

    f32x8 zero;
    #pragma unroll
    for (int i = 0; i < 8; ++i) zero[i] = 0.0f;
    f32x8 acc[4][2];
    #pragma unroll
    for (int mi = 0; mi < 4; ++mi)
        #pragma unroll
        for (int ni = 0; ni < 2; ++ni) acc[mi][ni] = zero;

    Frag32B a[2][4], b[2][2];

    // A fragment: 16x32 bf16 tile; lanes 0-15 hold K 0..7 & 16..23,
    // lanes 16-31 hold K 8..15 & 24..31 (ISA 16-bit A layout).
    // B fragment: 32x16 tile; lane n holds column n, same K split as A.
    #define LOAD_TILE(buf, kk)                                              \
        do {                                                                \
            _Pragma("unroll")                                               \
            for (int mi = 0; mi < 4; ++mi) {                                \
                a[buf][mi].u[0] = *(const uint4*)(pa[mi] + (kk));           \
                a[buf][mi].u[1] = *(const uint4*)(pa[mi] + (kk) + 16);      \
            }                                                               \
            _Pragma("unroll")                                               \
            for (int ni = 0; ni < 2; ++ni) {                                \
                b[buf][ni].u[0] = *(const uint4*)(pb[ni] + (kk));           \
                b[buf][ni].u[1] = *(const uint4*)(pb[ni] + (kk) + 16);      \
            }                                                               \
        } while (0)

    #define WMMA_TILE(buf)                                                  \
        do {                                                                \
            _Pragma("unroll")                                               \
            for (int mi = 0; mi < 4; ++mi)                                  \
                _Pragma("unroll")                                           \
                for (int ni = 0; ni < 2; ++ni)                              \
                    acc[mi][ni] = __builtin_amdgcn_wmma_f32_16x16x32_bf16(  \
                        false, a[buf][mi].v, false, b[buf][ni].v,           \
                        (short)0, acc[mi][ni], false, false);               \
        } while (0)

    LOAD_TILE(0, 0);
    for (int k0 = 0; k0 < K; k0 += 64) {
        LOAD_TILE(1, k0 + 32);                  // overlap with WMMA batch 0
        if (k0 + 64 < K)
            __builtin_prefetch(pa[0] + k0 + 64, 0, 3);
        WMMA_TILE(0);
        if (k0 + 64 < K) LOAD_TILE(0, k0 + 64); // overlap with WMMA batch 1
        WMMA_TILE(1);
    }
    #undef LOAD_TILE
    #undef WMMA_TILE

    // C layout: VGPR r, lanes 0-15 -> row r, lanes 16-31 -> row r+8, col=lane&15
    #pragma unroll
    for (int ni = 0; ni < 2; ++ni) {
        int col = nBase + ni * 16 + laneLo;
        if (col >= N) continue;
        float bv = bias ? bias[col] : 0.0f;
        #pragma unroll
        for (int mi = 0; mi < 4; ++mi) {
            #pragma unroll
            for (int r = 0; r < 8; ++r) {
                int row = mBase + mi * 16 + laneHi * 8 + r;
                size_t o = (size_t)row * N + col;
                float x = acc[mi][ni][r] + bv;
                if (res) x += res[o];
                if (GELU) x = 0.5f * x * (1.0f + erff(x * 0.70710678118654752f));
                if (Cf) Cf[o] = x;
                if (Cb) Cb[o] = f32_to_bf16(x);
            }
        }
    }
}

// ---------------------------------------------------------------------------
// LayerNorm: f32 in -> bf16 out.  One wave per row, wave32 shfl reduction.
// ---------------------------------------------------------------------------
__global__ __launch_bounds__(256)
void k_layernorm(const float* __restrict__ x, const float* __restrict__ g,
                 const float* __restrict__ b, unsigned short* __restrict__ y,
                 int C, int nRows)
{
    int row = blockIdx.x * (blockDim.x >> 5) + (threadIdx.x >> 5);
    if (row >= nRows) return;
    int lane = threadIdx.x & 31;
    const float* rx = x + (size_t)row * C;
    float s = 0.f, ss = 0.f;
    for (int c = lane; c < C; c += 32) { float v = rx[c]; s += v; ss += v * v; }
    #pragma unroll
    for (int m = 16; m >= 1; m >>= 1) { s += __shfl_xor(s, m); ss += __shfl_xor(ss, m); }
    float mean = s / C;
    float var  = ss / C - mean * mean;
    float inv  = rsqrtf(var + 1e-5f);
    unsigned short* ry = y + (size_t)row * C;
    for (int c = lane; c < C; c += 32)
        ry[c] = f32_to_bf16((rx[c] - mean) * inv * g[c] + b[c]);
}

// ---------------------------------------------------------------------------
// Attention: softmax(Q K^T * scale) V.  One wave per (group, head).
// Q: [G*Jq, H*hd], K/V: [G*Jk, H*hd], all bf16; O bf16 [G*Jq, H*hd].
// ---------------------------------------------------------------------------
__global__ __launch_bounds__(128)
void k_attn(const unsigned short* __restrict__ Q,
            const unsigned short* __restrict__ Km,
            const unsigned short* __restrict__ Vm,
            unsigned short* __restrict__ O,
            int G, int Jq, int Jk, int hd, float scale)
{
    __shared__ float sK[4][256];
    __shared__ float sV[4][256];
    const int wave = threadIdx.x >> 5;
    const int lane = threadIdx.x & 31;
    const int widx = blockIdx.x * 4 + wave;
    const bool active = widx < G * HEADS;
    const int g = widx / HEADS;
    const int h = widx - g * HEADS;
    const int C = HEADS * hd;

    if (active) {
        for (int idx = lane; idx < Jk * hd; idx += 32) {
            int j = idx / hd, d = idx - j * hd;
            size_t o = (size_t)(g * Jk + j) * C + h * hd + d;
            sK[wave][idx] = bf16_to_f32(Km[o]);
            sV[wave][idx] = bf16_to_f32(Vm[o]);
        }
    }
    __syncthreads();
    if (!active || lane >= Jq) return;

    float q[24];
    for (int d = 0; d < hd; ++d)
        q[d] = bf16_to_f32(Q[(size_t)(g * Jq + lane) * C + h * hd + d]);

    float s[16];
    float mx = -1e30f;
    for (int j = 0; j < Jk; ++j) {
        float a = 0.f;
        for (int d = 0; d < hd; ++d) a += q[d] * sK[wave][j * hd + d];
        a *= scale; s[j] = a; mx = fmaxf(mx, a);
    }
    float sum = 0.f;
    for (int j = 0; j < Jk; ++j) { s[j] = __expf(s[j] - mx); sum += s[j]; }
    float inv = 1.0f / sum;
    for (int d = 0; d < hd; ++d) {
        float a = 0.f;
        for (int j = 0; j < Jk; ++j) a += s[j] * sV[wave][j * hd + d];
        O[(size_t)(g * Jq + lane) * C + h * hd + d] = f32_to_bf16(a * inv);
    }
}

// ---------------------------------------------------------------------------
// Elementwise / data-movement kernels
// ---------------------------------------------------------------------------
__global__ void k_f32_to_bf16(unsigned short* __restrict__ d,
                              const float* __restrict__ s, long long n)
{
    long long i = blockIdx.x * (long long)blockDim.x + threadIdx.x;
    if (i < n) d[i] = f32_to_bf16(s[i]);
}

__global__ void k_gather_tokens(unsigned short* __restrict__ dst,
                                const float* __restrict__ x, int J, int useUp)
{
    long long total = (long long)B_DIM * T_DIM * J * CIN;
    long long idx = blockIdx.x * (long long)blockDim.x + threadIdx.x;
    if (idx >= total) return;
    int c = (int)(idx % CIN);
    long long r = idx / CIN;
    int j = (int)(r % J); r /= J;
    int t = (int)(r % T_DIM);
    int b = (int)(r / T_DIM);
    int sj = useUp ? c_UP[j] : c_DOWN[j];
    float v = x[(((size_t)b * CIN + c) * T_DIM + t) * P_DIM + sj];
    dst[idx] = f32_to_bf16(v);
}

// dst[(g*dstStride + dstOff + r)*C + c] = src[(g*srcStride + srcOff + r)*C + c]
__global__ void k_copy_rows_f2f(float* __restrict__ dst, const float* __restrict__ src,
                                int nG, int rows, int C, int dS, int dO, int sS, int sO)
{
    long long total = (long long)nG * rows * C;
    long long idx = blockIdx.x * (long long)blockDim.x + threadIdx.x;
    if (idx >= total) return;
    int c = (int)(idx % C);
    long long r = idx / C;
    int rr = (int)(r % rows);
    int g = (int)(r / rows);
    dst[((size_t)g * dS + dO + rr) * C + c] = src[((size_t)g * sS + sO + rr) * C + c];
}
__global__ void k_copy_rows_f2b(unsigned short* __restrict__ dst, const float* __restrict__ src,
                                int nG, int rows, int C, int dS, int dO, int sS, int sO)
{
    long long total = (long long)nG * rows * C;
    long long idx = blockIdx.x * (long long)blockDim.x + threadIdx.x;
    if (idx >= total) return;
    int c = (int)(idx % C);
    long long r = idx / C;
    int rr = (int)(r % rows);
    int g = (int)(r / rows);
    dst[((size_t)g * dS + dO + rr) * C + c] =
        f32_to_bf16(src[((size_t)g * sS + sO + rr) * C + c]);
}
__global__ void k_copy_rows_b2b(unsigned short* __restrict__ dst, const unsigned short* __restrict__ src,
                                int nG, int rows, int C, int dS, int dO, int sS, int sO)
{
    long long total = (long long)nG * rows * C;
    long long idx = blockIdx.x * (long long)blockDim.x + threadIdx.x;
    if (idx >= total) return;
    int c = (int)(idx % C);
    long long r = idx / C;
    int rr = (int)(r % rows);
    int g = (int)(r / rows);
    dst[((size_t)g * dS + dO + rr) * C + c] = src[((size_t)g * sS + sO + rr) * C + c];
}

// XF[g, v*256+c] : v<15 from Hl rows (g*15+v), else Hs rows (g*10+v-15)
__global__ void k_pack_xf(float* __restrict__ XF, const float* __restrict__ Hl,
                          const float* __restrict__ Hs)
{
    long long total = (long long)GS * FFD;
    long long idx = blockIdx.x * (long long)blockDim.x + threadIdx.x;
    if (idx >= total) return;
    int c = (int)(idx % 256);
    int v = (int)((idx / 256) % P_DIM);
    int g = (int)(idx / FFD);
    XF[idx] = (v < JL) ? Hl[((size_t)g * JL + v) * 256 + c]
                       : Hs[((size_t)g * JS + (v - JL)) * 256 + c];
}

// out[b,c,t,v] = Y[(b*T+t)*6400 + v*256 + c]
__global__ void k_final_out(float* __restrict__ out, const float* __restrict__ Y)
{
    long long total = (long long)B_DIM * 256 * T_DIM * P_DIM;
    long long idx = blockIdx.x * (long long)blockDim.x + threadIdx.x;
    if (idx >= total) return;
    int v = (int)(idx % P_DIM);
    long long r = idx / P_DIM;
    int t = (int)(r % T_DIM); r /= T_DIM;
    int c = (int)(r % 256);
    int b = (int)(r / 256);
    out[idx] = Y[((size_t)(b * T_DIM + t)) * FFD + v * 256 + c];
}

// ---------------------------------------------------------------------------
// Host-side orchestration
// ---------------------------------------------------------------------------
// Param indices: dict keys flattened in JAX sorted order (x1, x2, then params)
enum {
    P_X1 = 0, P_X2 = 1,
    P_CONV_L1_B = 2, P_CONV_L1_W, P_CONV_L2_B, P_CONV_L2_W,
    P_CONV_S1_B, P_CONV_S1_W, P_CONV_S2_B, P_CONV_S2_W,
    P_CRL_LN_B, P_CRL_LN_G, P_CRL_TOK_W, P_CRL_TOOUT_B, P_CRL_TOOUT_W, P_CRL_TOQ_W, P_CRL_TOV_W,
    P_CRS_LN_B, P_CRS_LN_G, P_CRS_TOK_W, P_CRS_TOOUT_B, P_CRS_TOOUT_W, P_CRS_TOQ_W, P_CRS_TOV_W,
    P_EL_FC1_B, P_EL_FC1_W, P_EL_FC2_B, P_EL_FC2_W, P_EL_LN2_B, P_EL_LN2_G,
    P_EL_LNK_B, P_EL_LNK_G, P_EL_LNQ_B, P_EL_LNQ_G, P_EL_LNV_B, P_EL_LNV_G,
    P_EL_PROJ_B, P_EL_PROJ_W, P_EL_WK, P_EL_WQ, P_EL_WV,
    P_ES_FC1_B, P_ES_FC1_W, P_ES_FC2_B, P_ES_FC2_W, P_ES_LN2_B, P_ES_LN2_G,
    P_ES_LNK_B, P_ES_LNK_G, P_ES_LNQ_B, P_ES_LNQ_G, P_ES_LNV_B, P_ES_LNV_G,
    P_ES_PROJ_B, P_ES_PROJ_W, P_ES_WK, P_ES_WQ, P_ES_WV,
    P_FLS_B, P_FLS_W, P_FSL_B, P_FSL_W,
    P_FF1_B, P_FF1_W, P_FF2_B, P_FF2_W, P_FFLN_B, P_FFLN_G,
    P_GLS_B, P_GLS_W, P_GSL_B, P_GSL_W,
    P_HL_B, P_HL_W, P_HLLN_B, P_HLLN_G,
    P_HS_B, P_HS_W, P_HSLN_B, P_HSLN_G
};

static inline long long cdiv(long long a, long long b) { return (a + b - 1) / b; }

struct Arena {
    char* base; size_t off;
    template <typename T> T* take(size_t n) {
        T* p = (T*)(base + off);
        off = (off + n * sizeof(T) + 255) & ~(size_t)255;
        return p;
    }
};

static void gemm(hipStream_t s, const unsigned short* A, const unsigned short* W,
                 const float* bias, const float* res, float* Cf, unsigned short* Cb,
                 int M, int N, int K, bool gelu)
{
    dim3 grid((unsigned)cdiv(N, 128), (unsigned)(M / 128));
    if (gelu) k_gemm_bf16<true ><<<grid, dim3(256), 0, s>>>(A, W, bias, res, Cf, Cb, M, N, K);
    else      k_gemm_bf16<false><<<grid, dim3(256), 0, s>>>(A, W, bias, res, Cf, Cb, M, N, K);
}

extern "C" void kernel_launch(void* const* d_in, const int* in_sizes, int n_in,
                              void* d_out, int out_size, void* d_ws, size_t ws_size,
                              hipStream_t stream)
{
    (void)in_sizes; (void)n_in; (void)out_size; (void)ws_size;
    auto F = [&](int i) { return (const float*)d_in[i]; };

    Arena ws{(char*)d_ws, 0};
    typedef unsigned short bfb;

    // ---- bf16 weights -----------------------------------------------------
    bfb* w_cs1 = ws.take<bfb>((size_t)CS * CIN);
    bfb* w_cs2 = ws.take<bfb>((size_t)CS * CIN);
    bfb* w_cl1 = ws.take<bfb>((size_t)CL * CIN);
    bfb* w_cl2 = ws.take<bfb>((size_t)CL * CIN);
    bfb* w_es_wq = ws.take<bfb>((size_t)CS * CS);
    bfb* w_es_wk = ws.take<bfb>((size_t)CS * CS);
    bfb* w_es_wv = ws.take<bfb>((size_t)CS * CS);
    bfb* w_es_pj = ws.take<bfb>((size_t)CS * CS);
    bfb* w_es_f1 = ws.take<bfb>((size_t)FFS * CS);
    bfb* w_es_f2 = ws.take<bfb>((size_t)CS * FFS);
    bfb* w_el_wq = ws.take<bfb>((size_t)CL * CL);
    bfb* w_el_wk = ws.take<bfb>((size_t)CL * CL);
    bfb* w_el_wv = ws.take<bfb>((size_t)CL * CL);
    bfb* w_el_pj = ws.take<bfb>((size_t)CL * CL);
    bfb* w_el_f1 = ws.take<bfb>((size_t)FFL * CL);
    bfb* w_el_f2 = ws.take<bfb>((size_t)CL * FFL);
    bfb* w_crl_k = ws.take<bfb>((size_t)CS * CS);
    bfb* w_crl_v = ws.take<bfb>((size_t)CS * CS);
    bfb* w_crl_q = ws.take<bfb>((size_t)CS * CS);
    bfb* w_crl_o = ws.take<bfb>((size_t)CS * CS);
    bfb* w_crs_k = ws.take<bfb>((size_t)CL * CL);
    bfb* w_crs_v = ws.take<bfb>((size_t)CL * CL);
    bfb* w_crs_q = ws.take<bfb>((size_t)CL * CL);
    bfb* w_crs_o = ws.take<bfb>((size_t)CL * CL);
    bfb* w_fls = ws.take<bfb>((size_t)CS * CL);
    bfb* w_fsl = ws.take<bfb>((size_t)CL * CS);
    bfb* w_gsl = ws.take<bfb>((size_t)CL * CS);
    bfb* w_gls = ws.take<bfb>((size_t)CS * CL);
    bfb* w_hs  = ws.take<bfb>((size_t)256 * CS);
    bfb* w_hl  = ws.take<bfb>((size_t)256 * CL);
    bfb* w_ff1 = ws.take<bfb>((size_t)FFD * FFD);
    bfb* w_ff2 = ws.take<bfb>((size_t)FFD * FFD);

    // ---- activations ------------------------------------------------------
    bfb* ag_s1 = ws.take<bfb>((size_t)RS * CIN);
    bfb* ag_s2 = ws.take<bfb>((size_t)RS * CIN);
    bfb* ag_l1 = ws.take<bfb>((size_t)RL * CIN);
    bfb* ag_l2 = ws.take<bfb>((size_t)RL * CIN);
    float* ts_q = ws.take<float>((size_t)RS * CS);
    float* ts_k = ws.take<float>((size_t)RS * CS);
    float* tl_q = ws.take<float>((size_t)RL * CL);
    float* tl_k = ws.take<float>((size_t)RL * CL);
    bfb* s_qn = ws.take<bfb>((size_t)RS * CS);
    bfb* s_kn = ws.take<bfb>((size_t)RS * CS);
    bfb* s_vn = ws.take<bfb>((size_t)RS * CS);
    bfb* s_q  = ws.take<bfb>((size_t)RS * CS);
    bfb* s_k  = ws.take<bfb>((size_t)RS * CS);
    bfb* s_v  = ws.take<bfb>((size_t)RS * CS);
    bfb* s_o  = ws.take<bfb>((size_t)RS * CS);
    float* xs = ws.take<float>((size_t)RS * CS);
    bfb* s_h  = ws.take<bfb>((size_t)RS * CS);
    bfb* s_g  = ws.take<bfb>((size_t)RS * FFS);
    float* xs2 = ws.take<float>((size_t)RS * CS);
    bfb* l_qn = ws.take<bfb>((size_t)RL * CL);
    bfb* l_kn = ws.take<bfb>((size_t)RL * CL);
    bfb* l_vn = ws.take<bfb>((size_t)RL * CL);
    bfb* l_q  = ws.take<bfb>((size_t)RL * CL);
    bfb* l_k  = ws.take<bfb>((size_t)RL * CL);
    bfb* l_v  = ws.take<bfb>((size_t)RL * CL);
    bfb* l_o  = ws.take<bfb>((size_t)RL * CL);
    float* xl = ws.take<float>((size_t)RL * CL);
    bfb* l_h  = ws.take<bfb>((size_t)RL * CL);
    bfb* l_g  = ws.take<bfb>((size_t)RL * FFL);
    float* xl2 = ws.take<float>((size_t)RL * CL);
    // cross
    bfb* lcls = ws.take<bfb>((size_t)GS * CL);
    bfb* scls = ws.take<bfb>((size_t)GS * CS);
    float* calql = ws.take<float>((size_t)GS * CS);
    float* calqs = ws.take<float>((size_t)GS * CL);
    float* catl  = ws.take<float>((size_t)RS * CS);
    bfb* catln = ws.take<bfb>((size_t)RS * CS);
    bfb* c_k   = ws.take<bfb>((size_t)RS * CS);
    bfb* c_v   = ws.take<bfb>((size_t)RS * CS);
    bfb* qcls  = ws.take<bfb>((size_t)GS * CS);
    bfb* c_q   = ws.take<bfb>((size_t)GS * CS);
    bfb* o_cl  = ws.take<bfb>((size_t)GS * CS);
    float* calol_f = ws.take<float>((size_t)GS * CS);
    bfb*   calol_b = ws.take<bfb>((size_t)GS * CS);
    float* row0l = ws.take<float>((size_t)GS * CL);
    float* cats  = ws.take<float>((size_t)RL * CL);
    bfb* catsn = ws.take<bfb>((size_t)RL * CL);
    bfb* c_k2  = ws.take<bfb>((size_t)RL * CL);
    bfb* c_v2  = ws.take<bfb>((size_t)RL * CL);
    bfb* qcls2 = ws.take<bfb>((size_t)GS * CL);
    bfb* c_q2  = ws.take<bfb>((size_t)GS * CL);
    bfb* o_cs  = ws.take<bfb>((size_t)GS * CL);
    float* calos_f = ws.take<float>((size_t)GS * CL);
    bfb*   calos_b = ws.take<bfb>((size_t)GS * CL);
    float* row0s = ws.take<float>((size_t)GS * CS);
    // heads + big FFN
    bfb* ln_hs = ws.take<bfb>((size_t)RS * CS);
    bfb* ln_hl = ws.take<bfb>((size_t)RL * CL);
    float* hs_out = ws.take<float>((size_t)RS * 256);
    float* hl_out = ws.take<float>((size_t)RL * 256);
    float* xf  = ws.take<float>((size_t)GS * FFD);
    bfb* xfn   = ws.take<bfb>((size_t)GS * FFD);
    bfb* g1    = ws.take<bfb>((size_t)GS * FFD);
    float* yf  = ws.take<float>((size_t)GS * FFD);

    // ---- helpers ----------------------------------------------------------
    auto cvt = [&](bfb* dst, int pidx, long long n) {
        k_f32_to_bf16<<<(unsigned)cdiv(n, 256), 256, 0, stream>>>(dst, F(pidx), n);
    };
    auto ln = [&](const float* x, int gi, int bi, bfb* y, int C, int rows) {
        k_layernorm<<<(unsigned)cdiv(rows, 8), 256, 0, stream>>>(x, F(gi), F(bi), y, C, rows);
    };
    auto cpy_ff = [&](float* d, const float* s, int nG, int rows, int C, int dS, int dO, int sS, int sO) {
        long long n = (long long)nG * rows * C;
        k_copy_rows_f2f<<<(unsigned)cdiv(n, 256), 256, 0, stream>>>(d, s, nG, rows, C, dS, dO, sS, sO);
    };
    auto cpy_fb = [&](bfb* d, const float* s, int nG, int rows, int C, int dS, int dO, int sS, int sO) {
        long long n = (long long)nG * rows * C;
        k_copy_rows_f2b<<<(unsigned)cdiv(n, 256), 256, 0, stream>>>(d, s, nG, rows, C, dS, dO, sS, sO);
    };
    auto cpy_bb = [&](bfb* d, const bfb* s, int nG, int rows, int C, int dS, int dO, int sS, int sO) {
        long long n = (long long)nG * rows * C;
        k_copy_rows_b2b<<<(unsigned)cdiv(n, 256), 256, 0, stream>>>(d, s, nG, rows, C, dS, dO, sS, sO);
    };
    const float sc24 = 0.20412414523193150f;   // 24^-0.5
    const float sc8  = 0.35355339059327373f;   // 8^-0.5
    const unsigned attnGrid = (GS * HEADS) / 4;

    // ---- 1. convert weights ----------------------------------------------
    cvt(w_cs1, P_CONV_S1_W, CS * CIN);  cvt(w_cs2, P_CONV_S2_W, CS * CIN);
    cvt(w_cl1, P_CONV_L1_W, CL * CIN);  cvt(w_cl2, P_CONV_L2_W, CL * CIN);
    cvt(w_es_wq, P_ES_WQ, CS * CS); cvt(w_es_wk, P_ES_WK, CS * CS);
    cvt(w_es_wv, P_ES_WV, CS * CS); cvt(w_es_pj, P_ES_PROJ_W, CS * CS);
    cvt(w_es_f1, P_ES_FC1_W, (long long)FFS * CS); cvt(w_es_f2, P_ES_FC2_W, (long long)CS * FFS);
    cvt(w_el_wq, P_EL_WQ, CL * CL); cvt(w_el_wk, P_EL_WK, CL * CL);
    cvt(w_el_wv, P_EL_WV, CL * CL); cvt(w_el_pj, P_EL_PROJ_W, CL * CL);
    cvt(w_el_f1, P_EL_FC1_W, (long long)FFL * CL); cvt(w_el_f2, P_EL_FC2_W, (long long)CL * FFL);
    cvt(w_crl_k, P_CRL_TOK_W, CS * CS); cvt(w_crl_v, P_CRL_TOV_W, CS * CS);
    cvt(w_crl_q, P_CRL_TOQ_W, CS * CS); cvt(w_crl_o, P_CRL_TOOUT_W, CS * CS);
    cvt(w_crs_k, P_CRS_TOK_W, CL * CL); cvt(w_crs_v, P_CRS_TOV_W, CL * CL);
    cvt(w_crs_q, P_CRS_TOQ_W, CL * CL); cvt(w_crs_o, P_CRS_TOOUT_W, CL * CL);
    cvt(w_fls, P_FLS_W, CS * CL); cvt(w_fsl, P_FSL_W, CL * CS);
    cvt(w_gsl, P_GSL_W, CL * CS); cvt(w_gls, P_GLS_W, CS * CL);
    cvt(w_hs, P_HS_W, 256 * CS);  cvt(w_hl, P_HL_W, 256 * CL);
    cvt(w_ff1, P_FF1_W, (long long)FFD * FFD);
    cvt(w_ff2, P_FF2_W, (long long)FFD * FFD);

    // ---- 2. gather tokens + conv ------------------------------------------
    {
        long long nS = (long long)RS * CIN, nL = (long long)RL * CIN;
        k_gather_tokens<<<(unsigned)cdiv(nS, 256), 256, 0, stream>>>(ag_s1, F(P_X1), JS, 0);
        k_gather_tokens<<<(unsigned)cdiv(nS, 256), 256, 0, stream>>>(ag_s2, F(P_X2), JS, 0);
        k_gather_tokens<<<(unsigned)cdiv(nL, 256), 256, 0, stream>>>(ag_l1, F(P_X1), JL, 1);
        k_gather_tokens<<<(unsigned)cdiv(nL, 256), 256, 0, stream>>>(ag_l2, F(P_X2), JL, 1);
    }
    gemm(stream, ag_s1, w_cs1, F(P_CONV_S1_B), nullptr, ts_q, nullptr, RS, CS, CIN, false);
    gemm(stream, ag_s2, w_cs2, F(P_CONV_S2_B), nullptr, ts_k, nullptr, RS, CS, CIN, false);
    gemm(stream, ag_l1, w_cl1, F(P_CONV_L1_B), nullptr, tl_q, nullptr, RL, CL, CIN, false);
    gemm(stream, ag_l2, w_cl2, F(P_CONV_L2_B), nullptr, tl_k, nullptr, RL, CL, CIN, false);

    // ---- 3. spatial encoder (small) ---------------------------------------
    ln(ts_q, P_ES_LNQ_G, P_ES_LNQ_B, s_qn, CS, RS);
    ln(ts_k, P_ES_LNK_G, P_ES_LNK_B, s_kn, CS, RS);
    ln(ts_q, P_ES_LNV_G, P_ES_LNV_B, s_vn, CS, RS);
    gemm(stream, s_qn, w_es_wq, nullptr, nullptr, nullptr, s_q, RS, CS, CS, false);
    gemm(stream, s_kn, w_es_wk, nullptr, nullptr, nullptr, s_k, RS, CS, CS, false);
    gemm(stream, s_vn, w_es_wv, nullptr, nullptr, nullptr, s_v, RS, CS, CS, false);
    k_attn<<<attnGrid, 128, 0, stream>>>(s_q, s_k, s_v, s_o, GS, JS, JS, 24, sc24);
    gemm(stream, s_o, w_es_pj, F(P_ES_PROJ_B), ts_q, xs, nullptr, RS, CS, CS, false);
    ln(xs, P_ES_LN2_G, P_ES_LN2_B, s_h, CS, RS);
    gemm(stream, s_h, w_es_f1, F(P_ES_FC1_B), nullptr, nullptr, s_g, RS, FFS, CS, true);
    gemm(stream, s_g, w_es_f2, F(P_ES_FC2_B), xs, xs2, nullptr, RS, CS, FFS, false);

    // ---- 4. spatial encoder (large) ---------------------------------------
    ln(tl_q, P_EL_LNQ_G, P_EL_LNQ_B, l_qn, CL, RL);
    ln(tl_k, P_EL_LNK_G, P_EL_LNK_B, l_kn, CL, RL);
    ln(tl_q, P_EL_LNV_G, P_EL_LNV_B, l_vn, CL, RL);
    gemm(stream, l_qn, w_el_wq, nullptr, nullptr, nullptr, l_q, RL, CL, CL, false);
    gemm(stream, l_kn, w_el_wk, nullptr, nullptr, nullptr, l_k, RL, CL, CL, false);
    gemm(stream, l_vn, w_el_wv, nullptr, nullptr, nullptr, l_v, RL, CL, CL, false);
    k_attn<<<attnGrid, 128, 0, stream>>>(l_q, l_k, l_v, l_o, GS, JL, JL, 8, sc8);
    gemm(stream, l_o, w_el_pj, F(P_EL_PROJ_B), tl_q, xl, nullptr, RL, CL, CL, false);
    ln(xl, P_EL_LN2_G, P_EL_LN2_B, l_h, CL, RL);
    gemm(stream, l_h, w_el_f1, F(P_EL_FC1_B), nullptr, nullptr, l_g, RL, FFL, CL, true);
    gemm(stream, l_g, w_el_f2, F(P_EL_FC2_B), xl, xl2, nullptr, RL, CL, FFL, false);

    // ---- 5. cross-scale exchange ------------------------------------------
    cpy_fb(lcls, xl2, GS, 1, CL, 1, 0, JL, 0);           // large class tokens
    cpy_fb(scls, xs2, GS, 1, CS, 1, 0, JS, 0);           // small class tokens
    gemm(stream, lcls, w_fls, F(P_FLS_B), nullptr, calql, nullptr, GS, CS, CL, false);
    gemm(stream, scls, w_fsl, F(P_FSL_B), nullptr, calqs, nullptr, GS, CL, CS, false);

    // cross_l: concat [cal_q_l, x_small]
    cpy_ff(catl, calql, GS, 1, CS, JS, 0, 1, 0);
    cpy_ff(catl, xs2,  GS, JS - 1, CS, JS, 1, JS, 1);
    ln(catl, P_CRL_LN_G, P_CRL_LN_B, catln, CS, RS);
    gemm(stream, catln, w_crl_k, nullptr, nullptr, nullptr, c_k, RS, CS, CS, false);
    gemm(stream, catln, w_crl_v, nullptr, nullptr, nullptr, c_v, RS, CS, CS, false);
    cpy_bb(qcls, catln, GS, 1, CS, 1, 0, JS, 0);
    gemm(stream, qcls, w_crl_q, nullptr, nullptr, nullptr, c_q, GS, CS, CS, false);
    k_attn<<<attnGrid, 128, 0, stream>>>(c_q, c_k, c_v, o_cl, GS, 1, JS, 24, sc24);
    gemm(stream, o_cl, w_crl_o, F(P_CRL_TOOUT_B), calql, calol_f, calol_b, GS, CS, CS, false);
    gemm(stream, calol_b, w_gsl, F(P_GSL_B), nullptr, row0l, nullptr, GS, CL, CS, false);
    cpy_ff(xl2, row0l, GS, 1, CL, JL, 0, 1, 0);          // xl row0 <- g_sl(cal_out)

    // cross_s: concat [cal_q_s, x_large] (rows 1..14 of original xl untouched)
    cpy_ff(cats, calqs, GS, 1, CL, JL, 0, 1, 0);
    cpy_ff(cats, xl2,  GS, JL - 1, CL, JL, 1, JL, 1);
    ln(cats, P_CRS_LN_G, P_CRS_LN_B, catsn, CL, RL);
    gemm(stream, catsn, w_crs_k, nullptr, nullptr, nullptr, c_k2, RL, CL, CL, false);
    gemm(stream, catsn, w_crs_v, nullptr, nullptr, nullptr, c_v2, RL, CL, CL, false);
    cpy_bb(qcls2, catsn, GS, 1, CL, 1, 0, JL, 0);
    gemm(stream, qcls2, w_crs_q, nullptr, nullptr, nullptr, c_q2, GS, CL, CL, false);
    k_attn<<<attnGrid, 128, 0, stream>>>(c_q2, c_k2, c_v2, o_cs, GS, 1, JL, 8, sc8);
    gemm(stream, o_cs, w_crs_o, F(P_CRS_TOOUT_B), calqs, calos_f, calos_b, GS, CL, CL, false);
    gemm(stream, calos_b, w_gls, F(P_GLS_B), nullptr, row0s, nullptr, GS, CS, CL, false);
    cpy_ff(xs2, row0s, GS, 1, CS, JS, 0, 1, 0);          // xs row0 <- g_ls(cal_out)

    // ---- 6. heads ---------------------------------------------------------
    ln(xs2, P_HSLN_G, P_HSLN_B, ln_hs, CS, RS);
    gemm(stream, ln_hs, w_hs, F(P_HS_B), nullptr, hs_out, nullptr, RS, 256, CS, false);
    ln(xl2, P_HLLN_G, P_HLLN_B, ln_hl, CL, RL);
    gemm(stream, ln_hl, w_hl, F(P_HL_B), nullptr, hl_out, nullptr, RL, 256, CL, false);

    // ---- 7. big FFN over flattened (v,c) ----------------------------------
    {
        long long n = (long long)GS * FFD;
        k_pack_xf<<<(unsigned)cdiv(n, 256), 256, 0, stream>>>(xf, hl_out, hs_out);
    }
    ln(xf, P_FFLN_G, P_FFLN_B, xfn, FFD, GS);
    gemm(stream, xfn, w_ff1, F(P_FF1_B), nullptr, nullptr, g1, GS, FFD, FFD, true);
    gemm(stream, g1,  w_ff2, F(P_FF2_B), xf, yf, nullptr, GS, FFD, FFD, false);

    // ---- 8. final layout transpose ----------------------------------------
    {
        long long n = (long long)B_DIM * 256 * T_DIM * P_DIM;
        k_final_out<<<(unsigned)cdiv(n, 256), 256, 0, stream>>>((float*)d_out, yf);
    }
}
